// RuleSynthesizer_9706626089003
// MI455X (gfx1250) — matrix operand order repaired and checked by reading, not compiled, not run on hardware
//
#include <hip/hip_runtime.h>
#include <hip/hip_bf16.h>
#include <math.h>

// ---------------- problem constants ----------------
#define NTOK 4096
#define ED   1024
#define SD   512
#define NP   512
#define MS   16
#define PD   256
#define NH   4
#define HD   64
#define SCALE 0.125f          // 1/sqrt(64)
#define INV_TEMP 1.25f        // 1/0.8

typedef __attribute__((ext_vector_type(16))) __bf16 bfx16;
typedef __attribute__((ext_vector_type(8)))  __bf16 bfx8;
typedef __attribute__((ext_vector_type(4)))  __bf16 bfx4;
typedef __attribute__((ext_vector_type(8)))  float  floatx8;

__device__ __forceinline__ float gelu_f(float x) {
    return 0.5f * x * (1.0f + erff(x * 0.70710678118654752f));
}

// ---------------- fp32 -> bf16 cast ----------------
__global__ void cast_bf16_kernel(const float* __restrict__ in, __bf16* __restrict__ out, int n) {
    int i = blockIdx.x * blockDim.x + threadIdx.x;
    int stride = gridDim.x * blockDim.x;
    for (; i < n; i += stride) out[i] = (__bf16)in[i];
}

// ---------------- WMMA GEMM:  C[M,Nn] = A[M,K](bf16) @ W[Nn,K]^T (bf16) + bias ----------------
// block: 256 threads = 8 waves; block tile 128(M) x 128(N); wave tile 32 x 64.
// Two A-fragments are reused against 4 B-fragments -> 8 WMMA per 32-K step per wave.
__global__ void gemm_wmma_kernel(const __bf16* __restrict__ A, const __bf16* __restrict__ W,
                                 const float* __restrict__ bias,
                                 float* __restrict__ Cf, __bf16* __restrict__ Cb,
                                 int M, int Nn, int K, int do_gelu) {
    const int tid  = threadIdx.x;
    const int wave = tid >> 5;
    const int lane = tid & 31;
    const int lh   = lane & 15;
    const int hi   = lane >> 4;        // 0 or 1 (half of the wave)
    const int wr   = wave & 3;         // wave row (32 rows each)
    const int wc   = wave >> 2;        // wave col (64 cols each)
    const int mb   = blockIdx.y * 128;
    const int nb   = blockIdx.x * 128;

    __shared__ alignas(16) __bf16 As[128 * 32];   // [row][k]
    __shared__ alignas(16) __bf16 Bs[128 * 32];   // [n][k]  (== W rows, contiguous K)

    floatx8 cacc[2][4] = {};

    const int lrow = tid >> 1, lhf = tid & 1;     // loader: 16 bf16 per thread per tile

    for (int k0 = 0; k0 < K; k0 += 32) {
        // ---- stage tiles into LDS ----
        const __bf16* ap = A + (size_t)(mb + lrow) * K + k0 + lhf * 16;
        *(bfx8*)(As + lrow * 32 + lhf * 16)     = *(const bfx8*)(ap);
        *(bfx8*)(As + lrow * 32 + lhf * 16 + 8) = *(const bfx8*)(ap + 8);
        const __bf16* wp = W + (size_t)(nb + lrow) * K + k0 + lhf * 16;
        *(bfx8*)(Bs + lrow * 32 + lhf * 16)     = *(const bfx8*)(wp);
        *(bfx8*)(Bs + lrow * 32 + lhf * 16 + 8) = *(const bfx8*)(wp + 8);
        if (k0 + 32 < K)
            __builtin_prefetch(ap + 32, 0, 1);
        __syncthreads();

        // ---- two A fragments (16x32 each, ISA layout) ----
        bfx16 a0, a1;
        ((bfx8*)&a0)[0] = *(const bfx8*)(As + (wr * 32 + lh) * 32 + hi * 8);
        ((bfx8*)&a0)[1] = *(const bfx8*)(As + (wr * 32 + lh) * 32 + hi * 8 + 16);
        ((bfx8*)&a1)[0] = *(const bfx8*)(As + (wr * 32 + 16 + lh) * 32 + hi * 8);
        ((bfx8*)&a1)[1] = *(const bfx8*)(As + (wr * 32 + 16 + lh) * 32 + hi * 8 + 16);

        // ---- 4 B fragments (32x16 each), each used twice ----
        #pragma unroll
        for (int j = 0; j < 4; ++j) {
            const int nc = wc * 64 + j * 16 + lh;
            bfx16 bfrag;
            ((bfx8*)&bfrag)[0] = *(const bfx8*)(Bs + nc * 32 + hi * 8);
            ((bfx8*)&bfrag)[1] = *(const bfx8*)(Bs + nc * 32 + hi * 8 + 16);
            cacc[0][j] = __builtin_amdgcn_wmma_f32_16x16x32_bf16(
                false, a0, false, bfrag, (short)0, cacc[0][j], false, false);
            cacc[1][j] = __builtin_amdgcn_wmma_f32_16x16x32_bf16(
                false, a1, false, bfrag, (short)0, cacc[1][j], false, false);
        }
        __syncthreads();
    }

    // ---- epilogue: bias (+gelu), fp32 and/or bf16 stores ----
    #pragma unroll
    for (int i = 0; i < 2; ++i) {
        #pragma unroll
        for (int j = 0; j < 4; ++j) {
            const int col = nb + wc * 64 + j * 16 + lh;
            const float bv = bias ? bias[col] : 0.0f;
            #pragma unroll
            for (int v = 0; v < 8; ++v) {
                const int row = mb + wr * 32 + i * 16 + hi * 8 + v;
                float val = cacc[i][j][v] + bv;
                if (do_gelu) val = gelu_f(val);
                if (Cf) Cf[(size_t)row * Nn + col] = val;
                if (Cb) Cb[(size_t)row * Nn + col] = (__bf16)val;
            }
        }
    }
}

// ---------------- flash attention ----------------
// grid: (NTOK/128, NH); block: 256 threads = 8 waves, each wave owns 16 query rows.
// K/V tiles (32 keys) staged once per block -> 64 WMMAs per stage.
// qkv: bf16 [NTOK, 3*PD] with q|k|v at col offsets 0|PD|2*PD, head h at +h*HD.
__global__ void flash_attn_kernel(const __bf16* __restrict__ qkv, __bf16* __restrict__ outb) {
    const int h     = blockIdx.y;
    const int qbase = blockIdx.x * 128;
    const int tid   = threadIdx.x;
    const int w     = tid >> 5;
    const int lane  = tid & 31;
    const int lh    = lane & 15;
    const int hi    = lane >> 4;

    __shared__ alignas(16) __bf16 Kt[32 * 64];     // [key][d]
    __shared__ alignas(16) __bf16 Vt[64 * 32];     // [d][key]  (transposed)
    __shared__ alignas(16) __bf16 Ps[8 * 16 * 32]; // per-wave P tile [row][key]

    // Q fragments (held for the whole K loop); A-frag layout 16x32 over d
    bfx16 q0, q1;
    {
        const __bf16* qp = qkv + (size_t)(qbase + w * 16 + lh) * (3 * PD) + h * HD;
        ((bfx8*)&q0)[0] = *(const bfx8*)(qp + hi * 16);
        ((bfx8*)&q0)[1] = *(const bfx8*)(qp + hi * 16 + 8);
        ((bfx8*)&q1)[0] = *(const bfx8*)(qp + 32 + hi * 16);
        ((bfx8*)&q1)[1] = *(const bfx8*)(qp + 32 + hi * 16 + 8);
    }

    floatx8 o0 = {}, o1 = {}, o2 = {}, o3 = {};   // 16 rows x 64 d (4 C tiles)
    float m[8], l[8];
    #pragma unroll
    for (int v = 0; v < 8; ++v) { m[v] = -1e30f; l[v] = 0.0f; }

    // cooperative loader mappings (256 threads)
    const int kkey = tid >> 3, kch = tid & 7;     // K tile: 8 bf16 per thread
    const int kp2  = tid >> 4, dc4 = tid & 15;    // V tile: 2 keys x 4 d per thread

    for (int kb = 0; kb < NTOK / 32; ++kb) {
        const int kbase = kb * 32;
        // ---- stage K (row-major) and V (transposed, packed-pair stores) ----
        {
            const __bf16* kp = qkv + (size_t)(kbase + kkey) * (3 * PD) + PD + h * HD + kch * 8;
            *(bfx8*)(Kt + kkey * 64 + kch * 8) = *(const bfx8*)(kp);
            const __bf16* vpa = qkv + (size_t)(kbase + 2 * kp2) * (3 * PD) + 2 * PD + h * HD + dc4 * 4;
            bfx4 va = *(const bfx4*)(vpa);
            bfx4 vb = *(const bfx4*)(vpa + 3 * PD);
            #pragma unroll
            for (int e = 0; e < 4; ++e) {
                union { __bf16 hh[2]; unsigned u; } pk;
                pk.hh[0] = va[e]; pk.hh[1] = vb[e];
                *(unsigned*)(Vt + (dc4 * 4 + e) * 32 + 2 * kp2) = pk.u;
            }
        }
        __syncthreads();

        // ---- S = Q * K^T   (2 key-subtiles x 2 d-steps) ----
        floatx8 s0 = {}, s1 = {};
        {
            bfx16 b;
            ((bfx8*)&b)[0] = *(const bfx8*)(Kt + lh * 64 + hi * 16);
            ((bfx8*)&b)[1] = *(const bfx8*)(Kt + lh * 64 + hi * 16 + 8);
            s0 = __builtin_amdgcn_wmma_f32_16x16x32_bf16(false, q0, false, b, (short)0, s0, false, false);
            ((bfx8*)&b)[0] = *(const bfx8*)(Kt + lh * 64 + 32 + hi * 16);
            ((bfx8*)&b)[1] = *(const bfx8*)(Kt + lh * 64 + 32 + hi * 16 + 8);
            s0 = __builtin_amdgcn_wmma_f32_16x16x32_bf16(false, q1, false, b, (short)0, s0, false, false);
            ((bfx8*)&b)[0] = *(const bfx8*)(Kt + (16 + lh) * 64 + hi * 16);
            ((bfx8*)&b)[1] = *(const bfx8*)(Kt + (16 + lh) * 64 + hi * 16 + 8);
            s1 = __builtin_amdgcn_wmma_f32_16x16x32_bf16(false, q0, false, b, (short)0, s1, false, false);
            ((bfx8*)&b)[0] = *(const bfx8*)(Kt + (16 + lh) * 64 + 32 + hi * 16);
            ((bfx8*)&b)[1] = *(const bfx8*)(Kt + (16 + lh) * 64 + 32 + hi * 16 + 8);
            s1 = __builtin_amdgcn_wmma_f32_16x16x32_bf16(false, q1, false, b, (short)0, s1, false, false);
        }

        // ---- online softmax (row stats live in the 16-lane column groups) ----
        #pragma unroll
        for (int v = 0; v < 8; ++v) {
            float a0 = s0[v] * SCALE, a1 = s1[v] * SCALE;
            float mx = fmaxf(a0, a1);
            #pragma unroll
            for (int off = 1; off < 16; off <<= 1) mx = fmaxf(mx, __shfl_xor(mx, off, 32));
            const float mn    = fmaxf(m[v], mx);
            const float alpha = __expf(m[v] - mn);
            const float p0 = __expf(a0 - mn), p1 = __expf(a1 - mn);
            float rs = p0 + p1;
            #pragma unroll
            for (int off = 1; off < 16; off <<= 1) rs += __shfl_xor(rs, off, 32);
            l[v] = l[v] * alpha + rs;
            m[v] = mn;
            o0[v] *= alpha; o1[v] *= alpha; o2[v] *= alpha; o3[v] *= alpha;
            const int r = hi * 8 + v;                      // C-layout row within the 16
            Ps[w * 512 + r * 32 + lh]      = (__bf16)p0;
            Ps[w * 512 + r * 32 + 16 + lh] = (__bf16)p1;   // DS in-order within wave
        }

        // ---- O += P * V ----
        bfx16 pa;
        ((bfx8*)&pa)[0] = *(const bfx8*)(Ps + w * 512 + lh * 32 + hi * 16);
        ((bfx8*)&pa)[1] = *(const bfx8*)(Ps + w * 512 + lh * 32 + hi * 16 + 8);
        {
            bfx16 vb;
            ((bfx8*)&vb)[0] = *(const bfx8*)(Vt + (0 * 16 + lh) * 32 + hi * 16);
            ((bfx8*)&vb)[1] = *(const bfx8*)(Vt + (0 * 16 + lh) * 32 + hi * 16 + 8);
            o0 = __builtin_amdgcn_wmma_f32_16x16x32_bf16(false, pa, false, vb, (short)0, o0, false, false);
            ((bfx8*)&vb)[0] = *(const bfx8*)(Vt + (1 * 16 + lh) * 32 + hi * 16);
            ((bfx8*)&vb)[1] = *(const bfx8*)(Vt + (1 * 16 + lh) * 32 + hi * 16 + 8);
            o1 = __builtin_amdgcn_wmma_f32_16x16x32_bf16(false, pa, false, vb, (short)0, o1, false, false);
            ((bfx8*)&vb)[0] = *(const bfx8*)(Vt + (2 * 16 + lh) * 32 + hi * 16);
            ((bfx8*)&vb)[1] = *(const bfx8*)(Vt + (2 * 16 + lh) * 32 + hi * 16 + 8);
            o2 = __builtin_amdgcn_wmma_f32_16x16x32_bf16(false, pa, false, vb, (short)0, o2, false, false);
            ((bfx8*)&vb)[0] = *(const bfx8*)(Vt + (3 * 16 + lh) * 32 + hi * 16);
            ((bfx8*)&vb)[1] = *(const bfx8*)(Vt + (3 * 16 + lh) * 32 + hi * 16 + 8);
            o3 = __builtin_amdgcn_wmma_f32_16x16x32_bf16(false, pa, false, vb, (short)0, o3, false, false);
        }
        __syncthreads();
    }

    // ---- normalize and store bf16 output [NTOK, PD] ----
    #pragma unroll
    for (int v = 0; v < 8; ++v) {
        const int row = qbase + w * 16 + hi * 8 + v;
        const float inv = 1.0f / l[v];
        __bf16* op = outb + (size_t)row * PD + h * HD + lh;
        op[0]  = (__bf16)(o0[v] * inv);
        op[16] = (__bf16)(o1[v] * inv);
        op[32] = (__bf16)(o2[v] * inv);
        op[48] = (__bf16)(o3[v] * inv);
    }
}

// ---------------- row LayerNorm (+residual, +optional gelu), 256 cols ----------------
__global__ void ln_kernel(const float* __restrict__ a, const float* __restrict__ res,
                          const float* __restrict__ g, const float* __restrict__ be,
                          float* __restrict__ outf, __bf16* __restrict__ outb, int do_gelu) {
    const int row = blockIdx.x, c = threadIdx.x;
    __shared__ float red[PD];
    float s = a[(size_t)row * PD + c];
    if (res) s += res[(size_t)row * PD + c];
    red[c] = s; __syncthreads();
    for (int off = 128; off > 0; off >>= 1) { if (c < off) red[c] += red[c + off]; __syncthreads(); }
    const float mean = red[0] * (1.0f / PD);
    __syncthreads();
    const float d = s - mean;
    red[c] = d * d; __syncthreads();
    for (int off = 128; off > 0; off >>= 1) { if (c < off) red[c] += red[c + off]; __syncthreads(); }
    const float var = red[0] * (1.0f / PD);
    float y = d * rsqrtf(var + 1e-5f) * g[c] + be[c];
    if (do_gelu) y = gelu_f(y);
    outf[(size_t)row * PD + c] = y;
    outb[(size_t)row * PD + c] = (__bf16)y;
}

// ---------------- pat = mean over rows (2-stage, deterministic) ----------------
__global__ void colmean_part_kernel(const float* __restrict__ x, float* __restrict__ part) {
    const int c = threadIdx.x, b = blockIdx.x;       // 32 blocks x 256 threads
    float s = 0.0f;
    const int r0 = b * (NTOK / 32);
    for (int r = 0; r < NTOK / 32; ++r) s += x[(size_t)(r0 + r) * PD + c];
    part[(size_t)b * PD + c] = s;
}
__global__ void colmean_fin_kernel(const float* __restrict__ part, float* __restrict__ pat,
                                   float* __restrict__ out_pat) {
    const int c = threadIdx.x;
    float s = 0.0f;
    for (int b = 0; b < 32; ++b) s += part[(size_t)b * PD + c];
    s *= (1.0f / NTOK);
    pat[c] = s;
    out_pat[c] = s;
}

// ---------------- head_input = [pat, state @ state_W^T + b] ----------------
__global__ void statectx_kernel(const float* __restrict__ pat, const float* __restrict__ state,
                                const float* __restrict__ sW, const float* __restrict__ sb,
                                float* __restrict__ hin) {
    const int j = threadIdx.x;       // 256
    hin[j] = pat[j];
    float acc = sb[j];
    const float* w = sW + (size_t)j * SD;
    for (int k = 0; k < SD; ++k) acc = fmaf(w[k], state[k], acc);
    hin[PD + j] = acc;
}

// ---------------- 16 tiny head MLPs + gumbel softmax ----------------
__global__ void heads_kernel(const float* __restrict__ hin,
                             const float* __restrict__ W1, const float* __restrict__ b1,
                             const float* __restrict__ W2, const float* __restrict__ b2,
                             const float* __restrict__ gum, float* __restrict__ prog) {
    const int hH = blockIdx.x, t = threadIdx.x;  // 16 blocks x 128 threads
    __shared__ float hbuf[128];
    __shared__ float lg[NP];
    __shared__ float red[128];
    float acc = b1[hH * 128 + t];
    const float* w = W1 + ((size_t)hH * 128 + t) * (2 * PD);
    for (int k = 0; k < 2 * PD; ++k) acc = fmaf(w[k], hin[k], acc);
    hbuf[t] = gelu_f(acc);
    __syncthreads();
    for (int n = t; n < NP; n += 128) {
        float a2 = b2[hH * NP + n];
        const float* w2 = W2 + ((size_t)hH * NP + n) * 128;
        for (int k = 0; k < 128; ++k) a2 = fmaf(w2[k], hbuf[k], a2);
        lg[n] = (a2 + gum[hH * NP + n]) * INV_TEMP;
    }
    __syncthreads();
    float mx = -1e30f;
    for (int n = t; n < NP; n += 128) mx = fmaxf(mx, lg[n]);
    red[t] = mx; __syncthreads();
    for (int off = 64; off > 0; off >>= 1) { if (t < off) red[t] = fmaxf(red[t], red[t + off]); __syncthreads(); }
    mx = red[0]; __syncthreads();
    float ss = 0.0f;
    for (int n = t; n < NP; n += 128) { float e = __expf(lg[n] - mx); lg[n] = e; ss += e; }
    red[t] = ss; __syncthreads();
    for (int off = 64; off > 0; off >>= 1) { if (t < off) red[t] += red[t + off]; __syncthreads(); }
    const float inv = 1.0f / red[0];
    for (int n = t; n < NP; n += 128) prog[(size_t)hH * NP + n] = lg[n] * inv;
}

// ---------------- stop gate ----------------
__global__ void stop_kernel(const float* __restrict__ pat, const float* __restrict__ state,
                            const float* __restrict__ W1, const float* __restrict__ b1,
                            const float* __restrict__ W2, const float* __restrict__ b2,
                            float* __restrict__ out) {
    const int t = threadIdx.x;   // 128
    __shared__ float red[128];
    float acc = b1[t];
    const float* w = W1 + (size_t)t * (PD + SD);
    for (int k = 0; k < PD; ++k) acc = fmaf(w[k], pat[k], acc);
    for (int k = 0; k < SD; ++k) acc = fmaf(w[PD + k], state[k], acc);
    red[t] = gelu_f(acc) * W2[t];
    __syncthreads();
    for (int off = 64; off > 0; off >>= 1) { if (t < off) red[t] += red[t + off]; __syncthreads(); }
    if (t == 0) {
        const float sp = 1.0f / (1.0f + __expf(-(red[0] + b2[0])));
        for (int i = 0; i < MS - 2; ++i) out[i] = sp;
    }
}

// ---------------- host orchestration ----------------
extern "C" void kernel_launch(void* const* d_in, const int* in_sizes, int n_in,
                              void* d_out, int out_size, void* d_ws, size_t ws_size,
                              hipStream_t stream) {
    const float* ex      = (const float*)d_in[0];
    const float* state   = (const float*)d_in[1];
    const float* proj_W  = (const float*)d_in[2];
    const float* proj_b  = (const float*)d_in[3];
    const float* proj_g  = (const float*)d_in[4];
    const float* proj_be = (const float*)d_in[5];
    const float* a1_in_w = (const float*)d_in[6];
    const float* a1_in_b = (const float*)d_in[7];
    const float* a1_o_w  = (const float*)d_in[8];
    const float* a1_o_b  = (const float*)d_in[9];
    const float* n1_g = (const float*)d_in[10], *n1_b = (const float*)d_in[11];
    const float* f1_W1 = (const float*)d_in[12], *f1_b1 = (const float*)d_in[13];
    const float* f1_W2 = (const float*)d_in[14], *f1_b2 = (const float*)d_in[15];
    const float* n2_g = (const float*)d_in[16], *n2_b = (const float*)d_in[17];
    const float* a2_in_w = (const float*)d_in[18], *a2_in_b = (const float*)d_in[19];
    const float* a2_o_w  = (const float*)d_in[20], *a2_o_b  = (const float*)d_in[21];
    const float* n3_g = (const float*)d_in[22], *n3_b = (const float*)d_in[23];
    const float* f2_W1 = (const float*)d_in[24], *f2_b1 = (const float*)d_in[25];
    const float* f2_W2 = (const float*)d_in[26], *f2_b2 = (const float*)d_in[27];
    const float* n4_g = (const float*)d_in[28], *n4_b = (const float*)d_in[29];
    const float* state_W = (const float*)d_in[30], *state_b = (const float*)d_in[31];
    const float* h_W1 = (const float*)d_in[32], *h_b1 = (const float*)d_in[33];
    const float* h_W2 = (const float*)d_in[34], *h_b2 = (const float*)d_in[35];
    const float* s_W1 = (const float*)d_in[36], *s_b1 = (const float*)d_in[37];
    const float* s_W2 = (const float*)d_in[38], *s_b2 = (const float*)d_in[39];
    const float* gum  = (const float*)d_in[40];

    float* out = (float*)d_out;              // [16*512 | 256 | 14]

    char* ws = (char*)d_ws;
    size_t off = 0;
    auto alloc = [&](size_t bytes) { void* p = ws + off; off = (off + bytes + 255) & ~(size_t)255; return p; };

    __bf16* exb    = (__bf16*)alloc((size_t)NTOK * ED * 2);
    __bf16* wproj  = (__bf16*)alloc((size_t)PD * ED * 2);
    __bf16* wa1in  = (__bf16*)alloc((size_t)3 * PD * PD * 2);
    __bf16* wa1o   = (__bf16*)alloc((size_t)PD * PD * 2);
    __bf16* wf1a   = (__bf16*)alloc((size_t)2 * PD * PD * 2);
    __bf16* wf1b   = (__bf16*)alloc((size_t)PD * 2 * PD * 2);
    __bf16* wa2in  = (__bf16*)alloc((size_t)3 * PD * PD * 2);
    __bf16* wa2o   = (__bf16*)alloc((size_t)PD * PD * 2);
    __bf16* wf2a   = (__bf16*)alloc((size_t)2 * PD * PD * 2);
    __bf16* wf2b   = (__bf16*)alloc((size_t)PD * 2 * PD * 2);
    float*  fbuf   = (float*)alloc((size_t)NTOK * 3 * PD * 4);  // widest gemm fp32 out
    float*  x      = (float*)alloc((size_t)NTOK * PD * 4);
    __bf16* xb     = (__bf16*)alloc((size_t)NTOK * PD * 2);
    __bf16* qkvb   = (__bf16*)alloc((size_t)NTOK * 3 * PD * 2);
    __bf16* attnb  = (__bf16*)alloc((size_t)NTOK * PD * 2);
    __bf16* ffnb   = (__bf16*)alloc((size_t)NTOK * 2 * PD * 2);
    float*  part   = (float*)alloc(32 * PD * 4);
    float*  pat    = (float*)alloc(PD * 4);
    float*  hin    = (float*)alloc(2 * PD * 4);

    auto cast = [&](const float* src, __bf16* dst, int n) {
        cast_bf16_kernel<<<1024, 256, 0, stream>>>(src, dst, n);
    };
    cast(ex, exb, NTOK * ED);
    cast(proj_W, wproj, PD * ED);
    cast(a1_in_w, wa1in, 3 * PD * PD);   cast(a1_o_w, wa1o, PD * PD);
    cast(f1_W1, wf1a, 2 * PD * PD);      cast(f1_W2, wf1b, PD * 2 * PD);
    cast(a2_in_w, wa2in, 3 * PD * PD);   cast(a2_o_w, wa2o, PD * PD);
    cast(f2_W1, wf2a, 2 * PD * PD);      cast(f2_W2, wf2b, PD * 2 * PD);

    auto gemm = [&](const __bf16* A, const __bf16* W, const float* bias,
                    float* Cf, __bf16* Cb, int M, int Nn, int K, int gelu) {
        gemm_wmma_kernel<<<dim3(Nn / 128, M / 128), 256, 0, stream>>>(A, W, bias, Cf, Cb, M, Nn, K, gelu);
    };
    auto ln = [&](const float* a, const float* res, const float* g, const float* be, int gelu) {
        ln_kernel<<<NTOK, PD, 0, stream>>>(a, res, g, be, x, xb, gelu);
    };
    auto attn = [&]() {
        flash_attn_kernel<<<dim3(NTOK / 128, NH), 256, 0, stream>>>(qkvb, attnb);
    };

    // x = gelu(ln(ex @ proj_W^T + b))
    gemm(exb, wproj, proj_b, fbuf, nullptr, NTOK, PD, ED, 0);
    ln(fbuf, nullptr, proj_g, proj_be, 1);

    // block 1: attention
    gemm(xb, wa1in, a1_in_b, nullptr, qkvb, NTOK, 3 * PD, PD, 0);
    attn();
    gemm(attnb, wa1o, a1_o_b, fbuf, nullptr, NTOK, PD, PD, 0);
    ln(x, fbuf, n1_g, n1_b, 0);
    // block 1: FFN
    gemm(xb, wf1a, f1_b1, nullptr, ffnb, NTOK, 2 * PD, PD, 1);
    gemm(ffnb, wf1b, f1_b2, fbuf, nullptr, NTOK, PD, 2 * PD, 0);
    ln(x, fbuf, n2_g, n2_b, 0);

    // block 2: attention
    gemm(xb, wa2in, a2_in_b, nullptr, qkvb, NTOK, 3 * PD, PD, 0);
    attn();
    gemm(attnb, wa2o, a2_o_b, fbuf, nullptr, NTOK, PD, PD, 0);
    ln(x, fbuf, n3_g, n3_b, 0);
    // block 2: FFN
    gemm(xb, wf2a, f2_b1, nullptr, ffnb, NTOK, 2 * PD, PD, 1);
    gemm(ffnb, wf2b, f2_b2, fbuf, nullptr, NTOK, PD, 2 * PD, 0);
    ln(x, fbuf, n4_g, n4_b, 0);

    // tails
    colmean_part_kernel<<<32, PD, 0, stream>>>(x, part);
    colmean_fin_kernel<<<1, PD, 0, stream>>>(part, pat, out + MS * NP);
    statectx_kernel<<<1, PD, 0, stream>>>(pat, state, state_W, state_b, hin);
    heads_kernel<<<MS, 128, 0, stream>>>(hin, h_W1, h_b1, h_W2, h_b2, gum, out);
    stop_kernel<<<1, 128, 0, stream>>>(pat, state, s_W1, s_b1, s_W2, s_b2, out + MS * NP + PD);
}
